// FFTOperations_17119739641966
// MI455X (gfx1250) — matrix operand order repaired and checked by reading, not compiled, not run on hardware
//
#include <hip/hip_runtime.h>
#include <hip/hip_bf16.h>
#include <math.h>

// ---------------------------------------------------------------------------
// FFT(32, Hann) -> |.| -> top-8 (stable ties) -> sparse real spectrum -> ifft.Re
//
// Memory-bound: 128 MB in + 128 MB out => ~11 us floor @ 23.3 TB/s.
// Forward DFT and reconstruction run as v_wmma_f32_16x16x32_f16 against
// constant cos/sin matrices. One wave32 handles a 16-row tile. The top-8
// selection is split across the lane pair (c, c+16): each lane extracts its
// own 16 bins' top-8 with sorting networks (2x Batcher sort-8 + bitonic
// combine), then a cross-lane bitonic merge of the two sorted-8 lists gives
// the global 8th-largest threshold: thresh = min_i max(a[i], b[7-i]).
// ---------------------------------------------------------------------------

typedef __attribute__((ext_vector_type(16))) _Float16 v16h;
typedef __attribute__((ext_vector_type(8)))  float    v8f;

#define WPB            8
#define BLOCK_THREADS  (WPB * 32)

// ds_swizzle_b32, group-of-32 mode: xor_mask=0x10, and_mask=0x1f
// => exchange with lane^16 (the partner lane holding the same column).
__device__ __forceinline__ float swapf(float x) {
  return __builtin_bit_cast(float,
      __builtin_amdgcn_ds_swizzle(__builtin_bit_cast(int, x), 0x401F));
}
__device__ __forceinline__ int swapi(int x) {
  return __builtin_amdgcn_ds_swizzle(x, 0x401F);
}

// Descending compare-exchange: a <- max, b <- min.
__device__ __forceinline__ void cswap(float& a, float& b) {
  const float hi = fmaxf(a, b);
  const float lo = fminf(a, b);
  a = hi; b = lo;
}

// Batcher odd-even merge sort, n=8, descending, 19 comparators.
__device__ __forceinline__ void sort8_desc(float v[8]) {
  cswap(v[0],v[1]); cswap(v[2],v[3]); cswap(v[4],v[5]); cswap(v[6],v[7]);
  cswap(v[0],v[2]); cswap(v[1],v[3]); cswap(v[4],v[6]); cswap(v[5],v[7]);
  cswap(v[1],v[2]); cswap(v[5],v[6]);
  cswap(v[0],v[4]); cswap(v[1],v[5]); cswap(v[2],v[6]); cswap(v[3],v[7]);
  cswap(v[2],v[4]); cswap(v[3],v[5]);
  cswap(v[1],v[2]); cswap(v[3],v[4]); cswap(v[5],v[6]);
}

__global__ __launch_bounds__(BLOCK_THREADS)
void fft_hann_topk8_recon(const float* __restrict__ x,
                          float* __restrict__ out,
                          int nrows, int ntiles)
{
  const int lane = threadIdx.x & 31;
  const int g    = lane >> 4;   // K-half selector (WMMA lane group)
  const int m    = lane & 15;   // A-matrix row / B,C,D column index

  const float w32 = 0.19634954084936207f; // 2*pi/32

  // --- Constant DFT matrices as WMMA A operands, contiguous k order (as B) -------------------
  // A element (M,K): lane = M + 16*(K>>4), half-slot i with
  // K = 16*(lane>>4) + i.
  v16h a_cos_lo, a_cos_hi, a_sin_lo, a_sin_hi;
  float win[16]; // Hann window for time samples n = 16*g + i (our B slots)
  #pragma unroll
  for (int i = 0; i < 16; ++i) {
    const int K  = (g << 4) + i;
    const float tl = w32 * (float)(m * K);
    const float th = w32 * (float)((m + 16) * K);
    a_cos_lo[i] = (_Float16)__cosf(tl);
    a_sin_lo[i] = (_Float16)__sinf(tl);
    a_cos_hi[i] = (_Float16)__cosf(th);
    a_sin_hi[i] = (_Float16)__sinf(th);
    const int n = (g << 4) + i;
    win[i] = 0.5f - 0.5f * __cosf(w32 * (float)n);
  }

  const int wave   = blockIdx.x * WPB + (threadIdx.x >> 5);
  const int nwaves = gridDim.x * WPB;

  for (int tile = wave; tile < ntiles; tile += nwaves) {
    const int  row   = tile * 16 + m;        // this lane's batch row (column)
    const bool valid = row < nrows;

    // --- Load 16 consecutive samples of one row: exactly the B layout -----
    const float* rp = x + (size_t)row * 32 + (g << 4);
    float4 f0 = {0,0,0,0}, f1 = {0,0,0,0}, f2 = {0,0,0,0}, f3 = {0,0,0,0};
    if (valid) {
      const float4* p4 = (const float4*)rp;
      f0 = p4[0]; f1 = p4[1]; f2 = p4[2]; f3 = p4[3];
    }
    // Prefetch next tile for this lane (gfx1250 global_prefetch_b8).
    {
      const long nrow = (long)(tile + nwaves) * 16 + m;
      if (nrow < (long)nrows)
        __builtin_prefetch(x + nrow * 32 + (g << 4), 0, 0);
    }

    float xs[16];
    xs[0]=f0.x; xs[1]=f0.y; xs[2]=f0.z;  xs[3]=f0.w;
    xs[4]=f1.x; xs[5]=f1.y; xs[6]=f1.z;  xs[7]=f1.w;
    xs[8]=f2.x; xs[9]=f2.y; xs[10]=f2.z; xs[11]=f2.w;
    xs[12]=f3.x;xs[13]=f3.y;xs[14]=f3.z; xs[15]=f3.w;

    v16h bm;
    #pragma unroll
    for (int i = 0; i < 16; ++i) bm[i] = (_Float16)(xs[i] * win[i]);

    // --- Forward DFT: 4 WMMAs (bins 0-15 / 16-31, Re / Im) ----------------
    v8f zero = {};
    v8f re_lo = __builtin_amdgcn_wmma_f32_16x16x32_f16(false, a_cos_lo, false, bm, (short)0, zero, false, false);
    v8f re_hi = __builtin_amdgcn_wmma_f32_16x16x32_f16(false, a_cos_hi, false, bm, (short)0, zero, false, false);
    v8f im_lo = __builtin_amdgcn_wmma_f32_16x16x32_f16(false, a_sin_lo, false, bm, (short)0, zero, false, false);
    v8f im_hi = __builtin_amdgcn_wmma_f32_16x16x32_f16(false, a_sin_hi, false, bm, (short)0, zero, false, false);

    // --- Own magnitudes (v_pk_fma + fast hardware v_sqrt): ----------------
    //   magLo[v] = bin 8g+v, magHi[v] = bin 16+8g+v
    float magLo[8], magHi[8];
    #pragma unroll
    for (int v = 0; v < 8; ++v) {
      magLo[v] = __builtin_amdgcn_sqrtf(__builtin_fmaf(re_lo[v], re_lo[v], im_lo[v]*im_lo[v]));
      magHi[v] = __builtin_amdgcn_sqrtf(__builtin_fmaf(re_hi[v], re_hi[v], im_hi[v]*im_hi[v]));
    }

    // --- Own top-8 (sorted desc) via sorting networks ---------------------
    float sLo[8], sHi[8];
    #pragma unroll
    for (int v = 0; v < 8; ++v) { sLo[v] = magLo[v]; sHi[v] = magHi[v]; }
    sort8_desc(sLo);
    sort8_desc(sHi);
    // Half-cleaner: top-8 of own 16 as a bitonic sequence...
    float t[8];
    #pragma unroll
    for (int j = 0; j < 8; ++j) t[j] = fmaxf(sLo[j], sHi[7 - j]);
    // ...then 3-stage bitonic merge sorts it descending (12 comparators).
    cswap(t[0],t[4]); cswap(t[1],t[5]); cswap(t[2],t[6]); cswap(t[3],t[7]);
    cswap(t[0],t[2]); cswap(t[1],t[3]); cswap(t[4],t[6]); cswap(t[5],t[7]);
    cswap(t[0],t[1]); cswap(t[2],t[3]); cswap(t[4],t[5]); cswap(t[6],t[7]);

    // --- Cross-lane bitonic merge with the partner's sorted-8 -------------
    // thresh = 8th largest of the full 32 = min_i max(t[i], tp[7-i]).
    float tp[8];
    #pragma unroll
    for (int j = 0; j < 8; ++j) tp[j] = swapf(t[j]);
    float thresh = fmaxf(t[0], tp[7]);
    #pragma unroll
    for (int j = 1; j < 8; ++j) thresh = fminf(thresh, fmaxf(t[j], tp[7 - j]));

    // --- Tie bookkeeping (jax top_k keeps equals in ascending-bin order) --
    int cntG = 0, eqLo = 0, eqHi = 0;
    #pragma unroll
    for (int v = 0; v < 8; ++v) {
      cntG += (magLo[v] > thresh) ? 1 : 0;
      cntG += (magHi[v] > thresh) ? 1 : 0;
      eqLo += (magLo[v] == thresh) ? 1 : 0;
      eqHi += (magHi[v] == thresh) ? 1 : 0;
    }
    const int pk  = cntG | (eqLo << 8) | (eqHi << 16);
    const int pkp = swapi(pk);
    const int cntGp = pkp & 0xff;
    const int eqLoP = (pkp >> 8) & 0xff;
    const int eqHiP = (pkp >> 16) & 0xff;

    const int allow = 8 - (cntG + cntGp);
    // Bin-order block layout: [0-7][8-15][16-23][24-31]; own blocks are
    // (8g..8g+7) and (16+8g..). Equal-count prefix entering each own block:
    const int baseLo = g ? eqLoP : 0;
    const int baseHi = g ? (eqLoP + eqLo + eqHiP) : (eqLo + eqLoP);

    // --- Keep decisions + scaled selected values for own bins -------------
    const float inv32 = 0.03125f;
    float selLo[8], selHi[8];
    int run = baseLo;
    #pragma unroll
    for (int v = 0; v < 8; ++v) {
      const float val  = magLo[v];
      const bool  isEq = (val == thresh);
      const bool  keep = (val > thresh) || (isEq && (run < allow));
      run += isEq ? 1 : 0;
      selLo[v] = keep ? val * inv32 : 0.0f;
    }
    run = baseHi;
    #pragma unroll
    for (int v = 0; v < 8; ++v) {
      const float val  = magHi[v];
      const bool  isEq = (val == thresh);
      const bool  keep = (val > thresh) || (isEq && (run < allow));
      run += isEq ? 1 : 0;
      selHi[v] = keep ? val * inv32 : 0.0f;
    }

    // --- Exchange the partner block this lane's B slots need --------------
    // g==0 needs bins 8-15  (= partner's own-lo block);  it sends its hi block.
    // g==1 needs bins 16-23 (= partner's own-hi block);  it sends its lo block.
    float recv[8];
    #pragma unroll
    for (int v = 0; v < 8; ++v)
      recv[v] = swapf(g ? selLo[v] : selHi[v]);

    // --- Sparse-spectrum B matrix: slot i holds bin K = 16g + i -----------
    v16h b2;
    #pragma unroll
    for (int i = 0; i < 8; ++i) {
      b2[i]     = (_Float16)(g ? recv[i]  : selLo[i]);  // bins 16g+0..7
      b2[i + 8] = (_Float16)(g ? selHi[i] : recv[i]);   // bins 16g+8..15
    }

    // --- Reconstruction: rec = cos-matrix @ sel (cos is symmetric) --------
    v8f rec_lo = __builtin_amdgcn_wmma_f32_16x16x32_f16(false, a_cos_lo, false, b2, (short)0, zero, false, false);
    v8f rec_hi = __builtin_amdgcn_wmma_f32_16x16x32_f16(false, a_cos_hi, false, b2, (short)0, zero, false, false);

    // --- Store: lane holds n = 8g+v and n = 16+8g+v of its row ------------
    if (valid) {
      float* op = out + (size_t)row * 32 + (g << 3);
      float4 s0 = {rec_lo[0], rec_lo[1], rec_lo[2], rec_lo[3]};
      float4 s1 = {rec_lo[4], rec_lo[5], rec_lo[6], rec_lo[7]};
      float4 s2 = {rec_hi[0], rec_hi[1], rec_hi[2], rec_hi[3]};
      float4 s3 = {rec_hi[4], rec_hi[5], rec_hi[6], rec_hi[7]};
      ((float4*)op)[0]        = s0;
      ((float4*)(op + 4))[0]  = s1;
      ((float4*)(op + 16))[0] = s2;
      ((float4*)(op + 20))[0] = s3;
    }
  }
}

extern "C" void kernel_launch(void* const* d_in, const int* in_sizes, int n_in,
                              void* d_out, int out_size, void* d_ws, size_t ws_size,
                              hipStream_t stream) {
  (void)n_in; (void)d_ws; (void)ws_size; (void)out_size;
  const float* x   = (const float*)d_in[0];
  float*       out = (float*)d_out;

  const int nrows  = in_sizes[0] / 32;          // B = 1,048,576
  const int ntiles = (nrows + 15) / 16;         // 16 rows per wave-tile

  int blocks = (ntiles + WPB - 1) / WPB;
  if (blocks > 4096) blocks = 4096;             // persistent: ~2 tiles/wave
  if (blocks < 1)    blocks = 1;

  fft_hann_topk8_recon<<<blocks, BLOCK_THREADS, 0, stream>>>(x, out, nrows, ntiles);
}
